// TFBlock_50165218018179
// MI455X (gfx1250) — matrix operand order repaired and checked by reading, not compiled, not run on hardware
//
#include <hip/hip_runtime.h>
#include <stdint.h>

// ---------------------------------------------------------------------------
// Types for CDNA5 WMMA (wave32): v_wmma_f32_16x16x32_bf16
// ---------------------------------------------------------------------------
typedef __bf16 bf16;
typedef __attribute__((ext_vector_type(8)))  __bf16 v8bf;
typedef __attribute__((ext_vector_type(16))) __bf16 v16bf;
typedef __attribute__((ext_vector_type(8)))  float  v8f;

#define DEV static __device__ __forceinline__

DEV v16bf cat16(v8bf lo, v8bf hi) {
  return __builtin_shufflevector(lo, hi, 0,1,2,3,4,5,6,7,8,9,10,11,12,13,14,15);
}
DEV v8bf ld8(const bf16* p) { return *(const v8bf*)p; }

// A-matrix 16x32 bf16 fragment (ISA 7.12.2): lane m = lane&15, kh = lane>>4
//   elems 0..7  hold K = kh*8 + j ; elems 8..15 hold K = 16 + kh*8 + j
DEV v16bf load_afrag(const bf16* row, int kh) {
  return cat16(ld8(row + kh * 8), ld8(row + 16 + kh * 8));
}
// B-matrix 32x16 bf16 fragment: lane n = lane&15, kh = lane>>4
//   elems 0..7  hold K = kh*8 + j ; elems 8..15 hold K = 16 + kh*8 + j
DEV v16bf load_bfrag(const bf16* col, int kh) {
  return cat16(ld8(col + kh * 8), ld8(col + 16 + kh * 8));
}
DEV v8f wmma_bf16(v16bf a, v16bf b, v8f c) {
  return __builtin_amdgcn_wmma_f32_16x16x32_bf16(false, a, false, b, (short)0, c,
                                                 false, false);
}
DEV v8f v8f_zero() { v8f z = {0.f,0.f,0.f,0.f,0.f,0.f,0.f,0.f}; return z; }

DEV uint32_t pack2(bf16 lo, bf16 hi) {
  return (uint32_t)__builtin_bit_cast(unsigned short, lo) |
         ((uint32_t)__builtin_bit_cast(unsigned short, hi) << 16);
}

// ---------------------------------------------------------------------------
// CDNA5 async global->LDS copy (ISA 15.18.3 op 98) + ASYNCcnt wait.
// LDS address for the VDST operand is the low 32 bits of the generic pointer
// (generic shared addr = SHARED_BASE aperture | lds_offset, ISA 10.2).
// ---------------------------------------------------------------------------
DEV void async_copy_b128(bf16* lds_dst, const bf16* gsrc) {
  asm volatile("global_load_async_to_lds_b128 %0, %1, off"
               :
               : "v"((uint32_t)(uintptr_t)lds_dst), "v"(gsrc)
               : "memory");
}
DEV void wait_async0() { asm volatile("s_wait_asynccnt 0x0" ::: "memory"); }

// ---------------------------------------------------------------------------
// fp32 -> bf16 conversion kernels (8-wide)
// ---------------------------------------------------------------------------
DEV v8bf pack8(float4 a, float4 b, float sc) {
  v8bf o = {(bf16)(a.x * sc), (bf16)(a.y * sc), (bf16)(a.z * sc), (bf16)(a.w * sc),
            (bf16)(b.x * sc), (bf16)(b.y * sc), (bf16)(b.z * sc), (bf16)(b.w * sc)};
  return o;
}

__global__ void f2bf_kernel(const float* __restrict__ s, bf16* __restrict__ d, int n) {
  int i = (blockIdx.x * blockDim.x + threadIdx.x) * 8;
  if (i < n) {
    float4 a = *(const float4*)(s + i);
    float4 b = *(const float4*)(s + i + 4);
    *(v8bf*)(d + i) = pack8(a, b, 1.0f);
  }
}

// Wq/Wk/Wv: src [H=16][D=1024][HS=64] -> dst [D][3*D] fused block (col offset
// chosen by caller), optional scale (1/sqrt(D) * log2(e) folded into Wq).
__global__ void wqkv_perm_kernel(const float* __restrict__ s, bf16* __restrict__ d,
                                 int ldn, float scale) {
  int i = (blockIdx.x * blockDim.x + threadIdx.x) * 8;
  if (i < 16 * 1024 * 64) {
    int hh = i >> 16;            // / (1024*64)
    int rem = i & 65535;
    int dd = rem >> 6;
    int ss = rem & 63;           // multiple of 8
    float4 a = *(const float4*)(s + i);
    float4 b = *(const float4*)(s + i + 4);
    *(v8bf*)(d + (size_t)dd * ldn + hh * 64 + ss) = pack8(a, b, scale);
  }
}

// ---------------------------------------------------------------------------
// Generic bf16 WMMA GEMM: C[M,N] = A[M,K] @ B[K,N] (+bias +resid, relu)
// Tile: 128x128 per WG (256 thr = 8 waves in 2x4), BK = 32 (one WMMA K-step),
// double-buffered LDS, async global->LDS for the A tile, dword-packed
// transpose for the B tile.
// ---------------------------------------------------------------------------
template <bool BIAS, bool RESID, bool RELU, bool OUTF, bool OUTB>
__global__ __launch_bounds__(256)
void gemm_bf16_kernel(const bf16* __restrict__ A, const bf16* __restrict__ B,
                      const float* __restrict__ bias, const float* __restrict__ resid,
                      float* __restrict__ outf, bf16* __restrict__ outb,
                      int M, int N, int K)
{
  constexpr int BM = 128, BN = 128, BK = 32, STR = 40;   // STR: +8 pad, 16B aligned
  __shared__ bf16 As[2][BM * STR];   // row-major  [m][k]
  __shared__ bf16 Bs[2][BN * STR];   // col-major  [n][k]

  const int t = threadIdx.x;
  const int lane = t & 31;
  const int wave = t >> 5;
  const int wm = wave >> 2;            // 0..1 : 64-row strip
  const int wn = wave & 3;             // 0..3 : 32-col strip
  const int lm = lane & 15, kh = lane >> 4;

  const int m0 = blockIdx.y * BM;
  const int n0 = blockIdx.x * BN;

  v8f acc[4][2];
  #pragma unroll
  for (int mt = 0; mt < 4; ++mt)
    #pragma unroll
    for (int nt = 0; nt < 2; ++nt) acc[mt][nt] = v8f_zero();

  const int arow = t >> 1, ahalf = t & 1;   // A stage: 128 rows x 2 halves of 16
  const int kp = t >> 4,  bseg = t & 15;    // B stage: 16 k-pairs x 16 segs of 8

  const bf16* Abase = A + (size_t)(m0 + arow) * K + ahalf * 16;
  const bf16* Bbase = B + (size_t)(2 * kp) * N + n0 + bseg * 8;

  v8bf br0, br1;
  auto a_async = [&](int k0, int buf) {
    bf16* dst = &As[buf][arow * STR + ahalf * 16];
    async_copy_b128(dst,     Abase + k0);
    async_copy_b128(dst + 8, Abase + k0 + 8);
  };
  auto b_fetch = [&](int k0) {
    const bf16* bg = Bbase + (size_t)k0 * N;
    br0 = ld8(bg);       // row k = k0 + 2*kp
    br1 = ld8(bg + N);   // row k+1
  };
  auto b_store = [&](int buf) {
    #pragma unroll
    for (int j = 0; j < 8; ++j)
      *(uint32_t*)&Bs[buf][(bseg * 8 + j) * STR + 2 * kp] = pack2(br0[j], br1[j]);
  };

  // prologue: stage tile 0
  a_async(0, 0);
  b_fetch(0);
  b_store(0);
  wait_async0();
  __syncthreads();

  int cur = 0;
  for (int k0 = 0; k0 < K; k0 += BK) {
    const bool more = (k0 + BK) < K;
    if (more) {
      a_async(k0 + BK, cur ^ 1);   // DMA next A tile while we compute
      b_fetch(k0 + BK);
    }

    v16bf af[4], bfr[2];
    #pragma unroll
    for (int mt = 0; mt < 4; ++mt)
      af[mt] = load_afrag(&As[cur][(wm * 64 + mt * 16 + lm) * STR], kh);
    #pragma unroll
    for (int nt = 0; nt < 2; ++nt)
      bfr[nt] = load_bfrag(&Bs[cur][(wn * 32 + nt * 16 + lm) * STR], kh);

    #pragma unroll
    for (int mt = 0; mt < 4; ++mt)
      #pragma unroll
      for (int nt = 0; nt < 2; ++nt)
        acc[mt][nt] = wmma_bf16(af[mt], bfr[nt], acc[mt][nt]);

    if (more) {
      b_store(cur ^ 1);
      wait_async0();
      __syncthreads();
      cur ^= 1;
    }
  }

  // Epilogue (branch-free). C/D layout: lanes 0-15 -> M=vr, 16-31 -> M=8+vr; N=lm
  #pragma unroll
  for (int mt = 0; mt < 4; ++mt) {
    #pragma unroll
    for (int nt = 0; nt < 2; ++nt) {
      const int gn = n0 + wn * 32 + nt * 16 + lm;
      float bv = 0.f;
      if (BIAS) bv = bias[gn];
      #pragma unroll
      for (int vr = 0; vr < 8; ++vr) {
        const int gm = m0 + wm * 64 + mt * 16 + kh * 8 + vr;
        const size_t o = (size_t)gm * N + gn;
        float v = acc[mt][nt][vr] + bv;
        if (RESID) v += resid[o];
        if (RELU)  v = fmaxf(v, 0.0f);
        if (OUTF)  outf[o] = v;
        if (OUTB)  outb[o] = (bf16)v;
      }
    }
  }
}

// ---------------------------------------------------------------------------
// Flash attention, causal. Scale AND log2(e) folded into Wq, so the scores are
// already in the exp2 domain: softmax probs = exp2(s - m) via v_exp_f32.
// Q/K/V: rows of the fused QKV buffer, row stride LD = 3*D.
// O: bf16 [B*T, D]. Grid: (T/64, H, B), block 128 (4 waves x 16 q-rows).
//  - Only the diagonal key tile (kt == qt) needs causal masking.
//  - Softmax row-sum comes free from the matrix pipe: V^T is augmented with a
//    ones-row (s=64), so a 5th accumulator column-tile carries the running
//    denominator, rescaled by corr exactly like O.
//  - Softmax max uses one tile-wide max (a valid monotone upper bound), so the
//    running max/corr are wave-uniform scalars; the rescale is skipped via a
//    uniform branch whenever the running max does not move.
// ---------------------------------------------------------------------------
__global__ __launch_bounds__(128)
void attention_kernel(const bf16* __restrict__ Qp, const bf16* __restrict__ Kp,
                      const bf16* __restrict__ Vp, bf16* __restrict__ O,
                      int T, int LD, int D)
{
  constexpr int HS = 64, KSTR = 72;   // 72*2B = 144B rows (16B aligned)
  __shared__ bf16 Ks[64 * KSTR];      // [u][s]       (B-frag layout for Q.K^T)
  __shared__ bf16 Vt[80 * KSTR];      // [s][u]; s=64 ones row, 65..79 zeros
  __shared__ bf16 Ps[4 * 16 * KSTR];  // per-wave P, [m][u]

  const int t = threadIdx.x, lane = t & 31, wave = t >> 5;
  const int lm = lane & 15, kh = lane >> 4;
  const int b = blockIdx.z, h = blockIdx.y, qt = blockIdx.x;
  const int q0 = qt * 64 + wave * 16;

  // one-time init of the augmented V^T rows (s = 64..79)
  for (int i = t; i < 16 * KSTR; i += 128) {
    const int srow = 64 + i / KSTR;
    Vt[(size_t)srow * KSTR + (i % KSTR)] = (srow == 64) ? (bf16)1.0f : (bf16)0.0f;
  }

  // Q fragments in registers (HS=64 -> two K=32 steps). A-layout row = lm.
  const bf16* qrow = Qp + (size_t)(b * T + q0 + lm) * LD + h * HS;
  v16bf aq[2];
  aq[0] = load_afrag(qrow, kh);
  aq[1] = load_afrag(qrow + 32, kh);

  v8f acc[5];                          // [0..3]: O tiles; [4]: denominator tile
  #pragma unroll
  for (int nt = 0; nt < 5; ++nt) acc[nt] = v8f_zero();
  float m = -__builtin_inff();         // wave-uniform running max (log2 domain)

  const int myrow = q0 + kh * 8;       // C-layout row base (add vr)
  const int kr = t >> 1, khalf = t & 1;   // K staging roles
  const int vpair = t >> 2, vseg = t & 3; // V staging roles (2 u-rows x 16 s)

  auto key_tile = [&](int u0, bool mask) {
    __syncthreads();
    {
      // K tile: straight row copy -> async DMA path
      const bf16* kg = Kp + (size_t)(b * T + u0 + kr) * LD + h * HS + khalf * 32;
      bf16* kl = &Ks[kr * KSTR + khalf * 32];
      async_copy_b128(kl,      kg);
      async_copy_b128(kl + 8,  kg + 8);
      async_copy_b128(kl + 16, kg + 16);
      async_copy_b128(kl + 24, kg + 24);
      // V tile: dword-packed transpose (rows u, u+1 -> one b32 per s)
      const bf16* vg = Vp + (size_t)(b * T + u0 + 2 * vpair) * LD + h * HS + vseg * 16;
      v8bf r0a = ld8(vg),      r0b = ld8(vg + 8);
      v8bf r1a = ld8(vg + LD), r1b = ld8(vg + LD + 8);
      #pragma unroll
      for (int j = 0; j < 8; ++j) {
        *(uint32_t*)&Vt[(vseg * 16 + j)     * KSTR + 2 * vpair] = pack2(r0a[j], r1a[j]);
        *(uint32_t*)&Vt[(vseg * 16 + 8 + j) * KSTR + 2 * vpair] = pack2(r0b[j], r1b[j]);
      }
      wait_async0();
    }
    __syncthreads();

    // S = Qs . K^T  (16 x 64 per wave, log2-domain scores)
    v8f s[4];
    #pragma unroll
    for (int nt = 0; nt < 4; ++nt) s[nt] = v8f_zero();
    #pragma unroll
    for (int kk = 0; kk < 2; ++kk)
      #pragma unroll
      for (int nt = 0; nt < 4; ++nt)
        s[nt] = wmma_bf16(aq[kk],
                          load_bfrag(&Ks[(nt * 16 + lm) * KSTR + kk * 32], kh),
                          s[nt]);

    // (diagonal-only) causal mask + per-lane tile max
    float pmax = -__builtin_inff();
    #pragma unroll
    for (int vr = 0; vr < 8; ++vr) {
      const int q = myrow + vr;
      #pragma unroll
      for (int nt = 0; nt < 4; ++nt) {
        float val = s[nt][vr];
        if (mask && (u0 + nt * 16 + lm > q)) val = -__builtin_inff();
        s[nt][vr] = val;
        pmax = fmaxf(pmax, val);
      }
    }
    // wave-wide max (upper bound for every row; exact softmax algebra)
    pmax = fmaxf(pmax, __shfl_xor(pmax, 1, 32));
    pmax = fmaxf(pmax, __shfl_xor(pmax, 2, 32));
    pmax = fmaxf(pmax, __shfl_xor(pmax, 4, 32));
    pmax = fmaxf(pmax, __shfl_xor(pmax, 8, 32));
    pmax = fmaxf(pmax, __shfl_xor(pmax, 16, 32));

    if (pmax > m) {                    // uniform branch: rescale only if needed
      const float corr = __builtin_amdgcn_exp2f(m - pmax);
      m = pmax;
      #pragma unroll
      for (int nt = 0; nt < 5; ++nt) acc[nt] = acc[nt] * corr;
    }
    #pragma unroll
    for (int vr = 0; vr < 8; ++vr)
      #pragma unroll
      for (int nt = 0; nt < 4; ++nt)
        Ps[(wave * 16 + kh * 8 + vr) * KSTR + nt * 16 + lm] =
            (bf16)__builtin_amdgcn_exp2f(s[nt][vr] - m);
    __syncthreads();

    // O += P . V_aug  (5th tile accumulates the softmax denominator)
    #pragma unroll
    for (int kk = 0; kk < 2; ++kk) {
      v16bf ap = load_afrag(&Ps[(wave * 16 + lm) * KSTR + kk * 32], kh);
      #pragma unroll
      for (int nt = 0; nt < 4; ++nt)
        acc[nt] = wmma_bf16(ap,
                            load_bfrag(&Vt[(nt * 16 + lm) * KSTR + kk * 32], kh),
                            acc[nt]);
      acc[4] = wmma_bf16(ap, load_bfrag(&Vt[(64 + lm) * KSTR + kk * 32], kh),
                         acc[4]);
    }
  };

  for (int kt = 0; kt < qt; ++kt) key_tile(kt * 64, false);  // interior: no mask
  key_tile(qt * 64, true);                                   // diagonal: masked

  // normalize: denominator lives in column 0 of tile 4 (lane kh*16)
  #pragma unroll
  for (int vr = 0; vr < 8; ++vr) {
    const float l = __shfl(acc[4][vr], kh * 16, 32);
    const float inv = 1.0f / l;
    #pragma unroll
    for (int nt = 0; nt < 4; ++nt)
      O[(size_t)(b * T + myrow + vr) * D + h * HS + nt * 16 + lm] =
          (bf16)(acc[nt][vr] * inv);
  }
}

// ---------------------------------------------------------------------------
// Host side
// ---------------------------------------------------------------------------
extern "C" void kernel_launch(void* const* d_in, const int* in_sizes, int n_in,
                              void* d_out, int out_size, void* d_ws, size_t ws_size,
                              hipStream_t stream) {
  (void)in_sizes; (void)n_in; (void)out_size; (void)ws_size;
  const int B = 4, T = 2048, D = 1024, DFF = 4096;
  const int BT = B * T;                       // 8192 rows
  const int N3 = 3 * D;                       // fused QKV width

  const float* x  = (const float*)d_in[0];
  const float* Wq = (const float*)d_in[1];
  const float* Wk = (const float*)d_in[2];
  const float* Wv = (const float*)d_in[3];
  const float* Wp = (const float*)d_in[4];
  const float* bp = (const float*)d_in[5];
  const float* W1 = (const float*)d_in[6];
  const float* b1 = (const float*)d_in[7];
  const float* W2 = (const float*)d_in[8];
  const float* b2 = (const float*)d_in[9];

  char* w = (char*)d_ws;
  auto take = [&](size_t bytes) -> char* {
    char* p = w;
    w += (bytes + 255) & ~(size_t)255;
    return p;
  };
  bf16* xb    = (bf16*)take((size_t)BT * D * 2);
  bf16* Wqkvb = (bf16*)take((size_t)D * N3 * 2);       // [D][3D] fused
  bf16* Wpb   = (bf16*)take((size_t)D * D * 2);
  bf16* W1b   = (bf16*)take((size_t)D * DFF * 2);
  bf16* W2b   = (bf16*)take((size_t)DFF * D * 2);
  bf16* qkvb  = (bf16*)take((size_t)BT * N3 * 2);      // [BT][3D] fused Q|K|V
  bf16* attnb = (bf16*)take((size_t)BT * D * 2);
  bf16* x1b   = (bf16*)take((size_t)BT * D * 2);
  float* x1f  = (float*)take((size_t)BT * D * 4);
  bf16* hb    = (bf16*)take((size_t)BT * DFF * 2);

  auto cvt = [&](const float* s, bf16* d, int n) {
    f2bf_kernel<<<(n / 8 + 255) / 256, 256, 0, stream>>>(s, d, n);
  };

  // 1) conversions / weight permutes.
  //    Wq carries 1/sqrt(D) * log2(e) so attention scores are exp2-domain.
  cvt(x, xb, BT * D);
  {
    const int g = (D * D / 8 + 255) / 256;
    const float qscale = 0.03125f * 1.44269504088896340736f;
    wqkv_perm_kernel<<<g, 256, 0, stream>>>(Wq, Wqkvb + 0 * D, N3, qscale);
    wqkv_perm_kernel<<<g, 256, 0, stream>>>(Wk, Wqkvb + 1 * D, N3, 1.0f);
    wqkv_perm_kernel<<<g, 256, 0, stream>>>(Wv, Wqkvb + 2 * D, N3, 1.0f);
  }
  cvt(Wp, Wpb, D * D);
  cvt(W1, W1b, D * DFF);
  cvt(W2, W2b, DFF * D);

  const dim3 blk(256);
  // 2) fused QKV projection: [8192,3072] = xb @ [1024,3072]
  {
    dim3 grid(N3 / 128, BT / 128);
    gemm_bf16_kernel<false, false, false, false, true>
        <<<grid, blk, 0, stream>>>(xb, Wqkvb, nullptr, nullptr, nullptr, qkvb,
                                   BT, N3, D);
  }
  // 3) causal flash attention -> attnb [B*T, D]
  {
    dim3 grid(T / 64, 16, B);
    attention_kernel<<<grid, dim3(128), 0, stream>>>(
        qkvb + 0 * D, qkvb + 1 * D, qkvb + 2 * D, attnb, T, N3, D);
  }
  // 4) x1 = x + attn @ Wp + bp   (fp32 + bf16 copies)
  {
    dim3 grid(D / 128, BT / 128);
    gemm_bf16_kernel<true, true, false, true, true>
        <<<grid, blk, 0, stream>>>(attnb, Wpb, bp, x, x1f, x1b, BT, D, D);
  }
  // 5) h = relu(x1 @ W1 + b1)
  {
    dim3 grid(DFF / 128, BT / 128);
    gemm_bf16_kernel<true, false, true, false, true>
        <<<grid, blk, 0, stream>>>(x1b, W1b, b1, nullptr, nullptr, hb,
                                   BT, DFF, D);
  }
  // 6) out = x1 + h @ W2 + b2   (fp32 final output)
  {
    dim3 grid(D / 128, BT / 128);
    gemm_bf16_kernel<true, true, false, true, false>
        <<<grid, blk, 0, stream>>>(hb, W2b, b2, x1f, (float*)d_out, nullptr,
                                   BT, D, DFF);
  }
}
